// SimpleMoE_52450140619161
// MI455X (gfx1250) — matrix operand order repaired and checked by reading, not compiled, not run on hardware
//
#include <hip/hip_runtime.h>
#include <hip/hip_bf16.h>
#include <stdint.h>

#define N_TOK 8192
#define DDIM  1024
#define HDIM  4096
#define ODIM  1024
#define NEXP  8

#define TM 32     // token tile (M) : two 16-row WMMA M-tiles, B fragments reused 2x
#define HC 256    // H chunk held in LDS per iteration

typedef __attribute__((ext_vector_type(16))) __bf16 v16bf;
typedef __attribute__((ext_vector_type(8)))  float  v8f;

__device__ __forceinline__ uint16_t f2bf(float f) {
  union { float f; uint32_t u; } a; a.f = f;
  uint32_t u = a.u;
  u += 0x7FFFu + ((u >> 16) & 1u);   // round-to-nearest-even
  return (uint16_t)(u >> 16);
}

// ---------------- x: fp32 -> bf16 (row-major unchanged) ----------------
__global__ void conv_x_kernel(const float* __restrict__ x, uint16_t* __restrict__ xb) {
  size_t i = ((size_t)blockIdx.x * blockDim.x + threadIdx.x) * 4;
  if (i >= (size_t)N_TOK * DDIM) return;
  float4 v = *(const float4*)(x + i);
  uint32_t lo = (uint32_t)f2bf(v.x) | ((uint32_t)f2bf(v.y) << 16);
  uint32_t hi = (uint32_t)f2bf(v.z) | ((uint32_t)f2bf(v.w) << 16);
  uint2 p; p.x = lo; p.y = hi;
  *(uint2*)(xb + i) = p;
}

// ------ weights: fp32 [E,K,N] -> bf16 WMMA B-fragment tiles ------
// Tile = 32(K) x 16(N). Fragment order: lane 0..31, 16 bf16 per lane.
// lanes 0-15: N=lane,    K=0..7,16..23 ; lanes 16-31: N=lane-16, K=8..15,24..31.
__global__ void conv_w_tiles(const float* __restrict__ W, uint16_t* __restrict__ Wt,
                             int Kdim, int Ndim) {
  __shared__ float tile[32][129];
  int ng = blockIdx.x;   // group of 128 N columns
  int kt = blockIdx.y;   // group of 32 K rows
  int e  = blockIdx.z;
  int t  = threadIdx.x;

  const float* src = W + ((size_t)e * Kdim + (size_t)kt * 32) * Ndim + (size_t)ng * 128;
  {
    int k  = t >> 3;
    int n0 = (t & 7) * 16;
    const float4* s4 = (const float4*)(src + (size_t)k * Ndim + n0);
#pragma unroll
    for (int q = 0; q < 4; ++q) {
      float4 v = s4[q];
      tile[k][n0 + q * 4 + 0] = v.x;
      tile[k][n0 + q * 4 + 1] = v.y;
      tile[k][n0 + q * 4 + 2] = v.z;
      tile[k][n0 + q * 4 + 3] = v.w;
    }
  }
  __syncthreads();
  int j    = t >> 5;          // which 16-col tile within the 128-col group
  int lane = t & 31;
  int nl   = j * 16 + (lane & 15);
  int kb   = (lane >> 4) * 16;
  size_t tileIdx = (((size_t)e * (Kdim / 32) + kt) * (size_t)(Ndim / 16) + (size_t)(ng * 8 + j));
  uint16_t* dst = Wt + tileIdx * 512 + (size_t)lane * 16;
#pragma unroll
  for (int i = 0; i < 16; ++i) dst[i] = f2bf(tile[(kb >> 1) + i + (i >= 8 ? 8 : 0)][nl]);
}

// ---------------- gate: scores, softmax, top-2, routing buckets ----------------
__global__ void gate_kernel(const float* __restrict__ x, const float* __restrict__ Wg,
                            const float* __restrict__ bg,
                            int* __restrict__ counts, int* __restrict__ rids,
                            float* __restrict__ rw) {
  int token = blockIdx.x * 8 + (threadIdx.x >> 5);
  int lane  = threadIdx.x & 31;
  const float* xr = x + (size_t)token * DDIM;

  float acc[NEXP];
#pragma unroll
  for (int e = 0; e < NEXP; ++e) acc[e] = 0.0f;

  for (int i = 0; i < DDIM / 32; ++i) {
    int d = i * 32 + lane;
    float xv = xr[d];
    const float4* wr = (const float4*)(Wg + (size_t)d * NEXP);
    float4 a = wr[0], b = wr[1];
    acc[0] += xv * a.x; acc[1] += xv * a.y; acc[2] += xv * a.z; acc[3] += xv * a.w;
    acc[4] += xv * b.x; acc[5] += xv * b.y; acc[6] += xv * b.z; acc[7] += xv * b.w;
  }
#pragma unroll
  for (int m = 16; m >= 1; m >>= 1) {
#pragma unroll
    for (int e = 0; e < NEXP; ++e) acc[e] += __shfl_xor(acc[e], m, 32);
  }
  if (lane == 0) {
    float s[NEXP], p[NEXP];
    float mx = -1e30f;
#pragma unroll
    for (int e = 0; e < NEXP; ++e) { s[e] = acc[e] + bg[e]; mx = fmaxf(mx, s[e]); }
    float sum = 0.0f;
#pragma unroll
    for (int e = 0; e < NEXP; ++e) { p[e] = __expf(s[e] - mx); sum += p[e]; }
    int i0 = 0; float b0 = s[0];
#pragma unroll
    for (int e = 1; e < NEXP; ++e) if (s[e] > b0) { b0 = s[e]; i0 = e; }
    int i1 = -1; float b1v = -1e30f;
#pragma unroll
    for (int e = 0; e < NEXP; ++e) if (e != i0 && s[e] > b1v) { b1v = s[e]; i1 = e; }
    float p0 = p[i0] / sum, p1 = p[i1] / sum;
    float inv = 1.0f / (p0 + p1 + 1e-8f);
    int pos0 = atomicAdd(&counts[i0], 1);
    rids[i0 * N_TOK + pos0] = token; rw[i0 * N_TOK + pos0] = p0 * inv;
    int pos1 = atomicAdd(&counts[i1], 1);
    rids[i1 * N_TOK + pos1] = token; rw[i1 * N_TOK + pos1] = p1 * inv;
  }
}

// ---------------- fragment loaders ----------------
// A 16x32 bf16 from LDS (row-major, rowStride elements):
// lane<16:  M=lane,     K = k0+{0..7}  and k0+16+{0..7}
// lane>=16: M=lane-16,  K = k0+8+{0..7} and k0+24+{0..7}
__device__ __forceinline__ v16bf load_a_frag(const uint16_t* s, int rowStride, int k0, int lane) {
  int row = lane & 15, sel = lane >> 4;
  const uint16_t* p = s + row * rowStride + k0 + sel * 8;
  union { uint4 q[2]; v16bf v; } u;
  u.q[0] = *(const uint4*)p;
  u.q[1] = *(const uint4*)(p + 16);
  return u.v;
}

// B 32x16 bf16 from pre-tiled global: 32 contiguous bytes per lane
__device__ __forceinline__ v16bf load_b_frag(const uint16_t* g) {
  union { uint4 q[2]; v16bf v; } u;
  u.q[0] = *(const uint4*)g;
  u.q[1] = *(const uint4*)(g + 8);
  return u.v;
}

// ---------------- fused expert FFN: relu(x@W1+b1)@W2+b2, weighted scatter ----------------
__global__ __launch_bounds__(256)
void moe_ffn_kernel(const uint16_t* __restrict__ xb,
                    const uint16_t* __restrict__ w1t,
                    const uint16_t* __restrict__ w2t,
                    const float* __restrict__ b1,
                    const float* __restrict__ b2,
                    const int* __restrict__ counts,
                    const int* __restrict__ rids,
                    const float* __restrict__ rw,
                    float* __restrict__ out) {
  __shared__ uint16_t xs[TM * DDIM];   // 64 KB gathered x tile (bf16)
  __shared__ uint16_t hs[TM * HC];     // 16 KB h chunk (bf16)
  __shared__ int   stok[TM];
  __shared__ float stw[TM];

  const int e     = blockIdx.y;
  const int cnt   = counts[e];
  const int start = blockIdx.x * TM;
  if (start >= cnt) return;

  const int tid  = threadIdx.x;
  const int lane = tid & 31;
  const int wv   = tid >> 5;           // 8 waves

  if (tid < TM) {
    int m = start + tid;
    stok[tid] = (m < cnt) ? rids[e * N_TOK + m] : 0;
    stw[tid]  = (m < cnt) ? rw[e * N_TOK + m] : 0.0f;   // pad rows contribute 0
  }
  __syncthreads();

  {   // gather 32 rows of bf16 x into LDS, 256B per thread
    int r  = tid >> 3;
    int c0 = (tid & 7) * 128;
    const uint4* src = (const uint4*)(xb + (size_t)stok[r] * DDIM + c0);
    uint4* dst = (uint4*)(xs + r * DDIM + c0);
#pragma unroll
    for (int q = 0; q < 16; ++q) dst[q] = src[q];
  }

  const v8f vzero = {};
  v8f acc2[2][8];                      // 2 M-tiles x 8 output column tiles per wave
#pragma unroll
  for (int mt = 0; mt < 2; ++mt)
#pragma unroll
    for (int j = 0; j < 8; ++j) acc2[mt][j] = vzero;

  for (int ch = 0; ch < HDIM / HC; ++ch) {      // 16 chunks of H
    __syncthreads();                            // prev chunk's phase-2 reads done

    // -------- phase 1: h_chunk = relu(x @ W1[:, chunk] + b1) --------
    v8f acc1[2][2];                             // 2 M-tiles x 2 column tiles per wave
#pragma unroll
    for (int mt = 0; mt < 2; ++mt)
#pragma unroll
      for (int j = 0; j < 2; ++j) acc1[mt][j] = vzero;

    for (int kt = 0; kt < DDIM / 32; ++kt) {
      v16bf A0 = load_a_frag(xs,             DDIM, kt * 32, lane);
      v16bf A1 = load_a_frag(xs + 16 * DDIM, DDIM, kt * 32, lane);
      const uint16_t* bb = w1t
          + ((((size_t)e * (DDIM / 32) + kt) * (HDIM / 16))
             + (size_t)ch * (HC / 16) + (size_t)wv * 2) * 512
          + (size_t)lane * 16;
      __builtin_prefetch(bb + (size_t)(HDIM / 16) * 512, 0, 1);  // next kt row of tiles
#pragma unroll
      for (int j = 0; j < 2; ++j) {
        v16bf B = load_b_frag(bb + j * 512);
        acc1[0][j] = __builtin_amdgcn_wmma_f32_16x16x32_bf16(
            false, A0, false, B, (short)0, acc1[0][j], false, false);
        acc1[1][j] = __builtin_amdgcn_wmma_f32_16x16x32_bf16(
            false, A1, false, B, (short)0, acc1[1][j], false, false);
      }
    }
    // bias + relu -> bf16 h chunk in LDS
#pragma unroll
    for (int mt = 0; mt < 2; ++mt) {
#pragma unroll
      for (int j = 0; j < 2; ++j) {
        int cl = wv * 32 + j * 16 + (lane & 15);
        float b1v = b1[(size_t)e * HDIM + (size_t)ch * HC + cl];
        int mrow = mt * 16 + (lane >> 4) * 8;
#pragma unroll
        for (int r = 0; r < 8; ++r) {
          float hval = acc1[mt][j][r] + b1v;
          hval = hval > 0.0f ? hval : 0.0f;
          hs[(mrow + r) * HC + cl] = f2bf(hval);
        }
      }
    }
    __syncthreads();

    // -------- phase 2: out += h_chunk @ W2[chunk, :] --------
    for (int kk = 0; kk < HC / 32; ++kk) {
      v16bf A0 = load_a_frag(hs,           HC, kk * 32, lane);
      v16bf A1 = load_a_frag(hs + 16 * HC, HC, kk * 32, lane);
      const uint16_t* bb = w2t
          + ((((size_t)e * (HDIM / 32) + (size_t)ch * (HC / 32) + kk) * (ODIM / 16))
             + (size_t)wv * 8) * 512
          + (size_t)lane * 16;
      __builtin_prefetch(bb + (size_t)(ODIM / 16) * 512, 0, 1);
#pragma unroll
      for (int j = 0; j < 8; ++j) {
        v16bf B = load_b_frag(bb + j * 512);
        acc2[0][j] = __builtin_amdgcn_wmma_f32_16x16x32_bf16(
            false, A0, false, B, (short)0, acc2[0][j], false, false);
        acc2[1][j] = __builtin_amdgcn_wmma_f32_16x16x32_bf16(
            false, A1, false, B, (short)0, acc2[1][j], false, false);
      }
    }
  }

  // -------- epilogue: + b2, * routing weight, scatter-add --------
#pragma unroll
  for (int mt = 0; mt < 2; ++mt) {
#pragma unroll
    for (int j = 0; j < 8; ++j) {
      int col = wv * 128 + j * 16 + (lane & 15);
      float b2v = b2[(size_t)e * ODIM + col];
      int mrow = mt * 16 + (lane >> 4) * 8;
#pragma unroll
      for (int r = 0; r < 8; ++r) {
        int m = mrow + r;
        float val = (acc2[mt][j][r] + b2v) * stw[m];
        atomicAdd(&out[(size_t)stok[m] * ODIM + col], val);
      }
    }
  }
}

// ---------------- host side ----------------
extern "C" void kernel_launch(void* const* d_in, const int* in_sizes, int n_in,
                              void* d_out, int out_size, void* d_ws, size_t ws_size,
                              hipStream_t stream) {
  const float* x  = (const float*)d_in[0];
  const float* Wg = (const float*)d_in[1];
  const float* bg = (const float*)d_in[2];
  const float* W1 = (const float*)d_in[3];
  const float* b1 = (const float*)d_in[4];
  const float* W2 = (const float*)d_in[5];
  const float* b2 = (const float*)d_in[6];
  float* out = (float*)d_out;

  char* ws = (char*)d_ws;
  int*      counts = (int*)(ws);
  int*      rids   = (int*)(ws + 256);
  float*    rw     = (float*)(ws + 256 + 4ull * NEXP * N_TOK);
  uint16_t* xbf    = (uint16_t*)(ws + 256 + 8ull * NEXP * N_TOK);
  uint16_t* w1t    = (uint16_t*)((char*)xbf + 2ull * N_TOK * DDIM);
  uint16_t* w2t    = (uint16_t*)((char*)w1t + 2ull * NEXP * DDIM * HDIM);

  hipMemsetAsync(d_out, 0, (size_t)out_size * sizeof(float), stream);
  hipMemsetAsync(counts, 0, 256, stream);

  conv_x_kernel<<<(N_TOK * DDIM / 4 + 255) / 256, 256, 0, stream>>>(x, xbf);
  conv_w_tiles<<<dim3(HDIM / 128, DDIM / 32, NEXP), 256, 0, stream>>>(W1, w1t, DDIM, HDIM);
  conv_w_tiles<<<dim3(ODIM / 128, HDIM / 32, NEXP), 256, 0, stream>>>(W2, w2t, HDIM, ODIM);
  gate_kernel<<<N_TOK / 8, 256, 0, stream>>>(x, Wg, bg, counts, rids, rw);
  moe_ffn_kernel<<<dim3(N_TOK / TM, NEXP), 256, 0, stream>>>(
      xbf, w1t, w2t, b1, b2, counts, rids, rw, out);
}
